// EX_Module_39032662786748
// MI455X (gfx1250) — compile-verified, matrix operand order checked by reading, not run on hardware
//
#include <hip/hip_runtime.h>
#include <hip/hip_bf16.h>

#define BB 32
#define CC 256
#define HW 4096
#define LN_EPS 1e-5f

typedef __attribute__((ext_vector_type(16))) __bf16 bf16x16;
typedef __attribute__((ext_vector_type(8)))  float  f32x8;
typedef __attribute__((ext_vector_type(4)))  unsigned int u32x4;

union Frag { bf16x16 v; u32x4 q[2]; };

__device__ __forceinline__ float hsig(float t) {
    return fminf(fmaxf(t * (1.0f / 6.0f) + 0.5f, 0.0f), 1.0f);
}

__device__ __forceinline__ unsigned short f2bf(float f) {
    unsigned int u = __float_as_uint(f);
    unsigned int r = u + 0x7FFFu + ((u >> 16) & 1u);
    return (unsigned short)(r >> 16);
}

__device__ __forceinline__ float waveSum(float v) {
#pragma unroll
    for (int o = 16; o > 0; o >>= 1) v += __shfl_xor(v, o, 32);
    return v;
}
__device__ __forceinline__ float waveMax(float v) {
#pragma unroll
    for (int o = 16; o > 0; o >>= 1) v = fmaxf(v, __shfl_xor(v, o, 32));
    return v;
}
__device__ __forceinline__ float blockSum256(float v, float* sbuf) {
    v = waveSum(v);
    int lane = threadIdx.x & 31, w = threadIdx.x >> 5;
    if (lane == 0) sbuf[w] = v;
    __syncthreads();
    float r = (threadIdx.x < 8) ? sbuf[threadIdx.x] : 0.0f;
    if (w == 0) { r = waveSum(r); if (lane == 0) sbuf[0] = r; }
    __syncthreads();
    r = sbuf[0];
    __syncthreads();
    return r;
}
__device__ __forceinline__ float blockMax256(float v, float* sbuf) {
    v = waveMax(v);
    int lane = threadIdx.x & 31, w = threadIdx.x >> 5;
    if (lane == 0) sbuf[w] = v;
    __syncthreads();
    float r = (threadIdx.x < 8) ? sbuf[threadIdx.x] : -1e30f;
    if (w == 0) { r = waveMax(r); if (lane == 0) sbuf[0] = r; }
    __syncthreads();
    r = sbuf[0];
    __syncthreads();
    return r;
}

// ---------------- K0: res_w fp32 -> bf16 -------------------------------
__global__ void k_cvt_resw(const float* __restrict__ rw, unsigned short* __restrict__ out) {
    int i = blockIdx.x * 256 + threadIdx.x;
    out[i] = f2bf(rw[i]);
}

// ---------------- K1: xmean[b,c] = mean_hw x ---------------------------
__global__ void k_xmean(const float* __restrict__ x, float* __restrict__ xmean) {
    __shared__ float sbuf[8];
    size_t base = (size_t)blockIdx.x * HW;
    const float4* x4 = (const float4*)(x + base);
    float s = 0.0f;
#pragma unroll
    for (int i = 0; i < 4; i++) {
        float4 v = x4[i * 256 + threadIdx.x];
        s += v.x + v.y + v.z + v.w;
    }
    s = blockSum256(s, sbuf);
    if (threadIdx.x == 0) xmean[blockIdx.x] = s * (1.0f / (float)HW);
}

// ---------------- K2: sa[b,hw] = conv1(x) ------------------------------
__global__ void k_sa(const float* __restrict__ x, const float* __restrict__ w1,
                     const float* __restrict__ b1, float* __restrict__ sa) {
    int b = blockIdx.y;
    int j = blockIdx.x * 256 + threadIdx.x;
    const float* xb = x + (size_t)b * CC * HW + j;
    float acc = b1[0];
#pragma unroll 4
    for (int c = 0; c < CC; c++) acc = fmaf(__ldg(&w1[c]), __ldg(&xb[(size_t)c * HW]), acc);
    sa[(size_t)b * HW + j] = acc;
}

// ---------------- K3: channel-attn vector path -> caL[b,c] -------------
__global__ void k_ca(const float* __restrict__ xmean,
                     const float* __restrict__ c0w, const float* __restrict__ c0b,
                     const float* __restrict__ b1w, const float* __restrict__ b1b,
                     const float* __restrict__ lng, const float* __restrict__ lnb,
                     float* __restrict__ caL) {
    __shared__ float xm[CC];
    __shared__ float caH[CC / 2];
    __shared__ float sbuf[8];
    int b = blockIdx.x, t = threadIdx.x;
    xm[t] = xmean[b * CC + t];
    __syncthreads();
    if (t < CC / 2) {
        float s = c0b[t];
        const float* row = c0w + t * CC;
#pragma unroll 8
        for (int c = 0; c < CC; c++) s = fmaf(row[c], xm[c], s);
        caH[t] = s;
    }
    __syncthreads();
    float s2 = b1b[t];
    {
        const float* row = b1w + t * (CC / 2);
#pragma unroll 8
        for (int i = 0; i < CC / 2; i++) s2 = fmaf(row[i], caH[i], s2);
    }
    float cav = hsig(s2);
    float mu = blockSum256(cav, sbuf) * (1.0f / (float)CC);
    float d = cav - mu;
    float var = blockSum256(d * d, sbuf) * (1.0f / (float)CC);
    float v = d * __frsqrt_rn(var + LN_EPS) * lng[t] + lnb[t];
    caL[b * CC + t] = v;
}

// ---------------- K4: u[b,c] = mean_hw(seq+par) ------------------------
__global__ void k_u(const float* __restrict__ sa, const float* __restrict__ caL,
                    float* __restrict__ u) {
    __shared__ float sbuf[8];
    int c = blockIdx.x, b = blockIdx.y;
    float ca = __ldg(&caL[b * CC + c]);
    const float* sab = sa + (size_t)b * HW;
    float acc = 0.0f;
#pragma unroll 4
    for (int i = 0; i < 16; i++) {
        float s = sab[i * 256 + threadIdx.x];
        acc += hsig(ca * s) + hsig(ca + s);
    }
    acc = blockSum256(acc, sbuf);
    if (threadIdx.x == 0) u[b * CC + c] = acc * (1.0f / (float)HW);
}

// ---------------- K5: SK matmuls + channel softmax -> a[b,c] -----------
__global__ void k_sk(const float* __restrict__ u,
                     const float* __restrict__ w1, const float* __restrict__ w2,
                     float* __restrict__ aArr) {
    __shared__ float uu[CC];
    __shared__ float uh[CC / 2];
    __shared__ float sbuf[8];
    int b = blockIdx.x, t = threadIdx.x;
    uu[t] = u[b * CC + t];
    __syncthreads();
    if (t < CC / 2) {
        float s = 0.0f;
        const float* row = w1 + t * CC;
#pragma unroll 8
        for (int c = 0; c < CC; c++) s = fmaf(row[c], uu[c], s);
        uh[t] = s;
    }
    __syncthreads();
    float s2 = 0.0f;
    {
        const float* row = w2 + t * (CC / 2);
#pragma unroll 8
        for (int i = 0; i < CC / 2; i++) s2 = fmaf(row[i], uh[i], s2);
    }
    float m = blockMax256(s2, sbuf);
    float e = __expf(s2 - m);
    float Z = blockSum256(e, sbuf);
    aArr[b * CC + t] = e / Z;
}

// ---------------- K6: Z[b,c], S[b,c] for spatial softmax ---------------
__global__ void k_zs(const float* __restrict__ x, const float* __restrict__ sa,
                     const float* __restrict__ caL, const float* __restrict__ aArr,
                     float* __restrict__ invZ, float* __restrict__ SoZ) {
    __shared__ float sbuf[8];
    int c = blockIdx.x, b = blockIdx.y;
    float ca = __ldg(&caL[b * CC + c]);
    float av = __ldg(&aArr[b * CC + c]);
    const float* sab = sa + (size_t)b * HW;
    const float* xbc = x + ((size_t)b * CC + c) * HW;
    float z = 0.0f, sx = 0.0f;
#pragma unroll 4
    for (int i = 0; i < 16; i++) {
        int j = i * 256 + threadIdx.x;
        float s = sab[j];
        float sk = av * hsig(ca * s) + (1.0f - av) * hsig(ca + s);
        float E = __expf(sk);
        z += E;
        sx = fmaf(xbc[j], E, sx);
    }
    z = blockSum256(z, sbuf);
    sx = blockSum256(sx, sbuf);
    if (threadIdx.x == 0) {
        float iz = 1.0f / z;
        invZ[b * CC + c] = iz;
        SoZ[b * CC + c] = sx * iz;
    }
}

// ---------------- K7: WMMA GEMM (res_w @ x) + fused epilogue -----------
// Per block: b fixed, hw tile of 64, full M=256.  8 waves, wave w -> rows
// [32w, 32w+32) as 2 M-tiles x 4 N-tiles of 16x16, K=256 in steps of 32.
#define LDSK 264   // 256 + 8 bf16 pad -> 528B row stride (16B aligned, conflict-free b128 reads)

__global__ void __launch_bounds__(256, 2)
k_gemm(const float* __restrict__ x, const unsigned short* __restrict__ reswbf,
       const float* __restrict__ resb, const float* __restrict__ sa,
       const float* __restrict__ caL, const float* __restrict__ aArr,
       const float* __restrict__ invZ, const float* __restrict__ SoZ,
       float* __restrict__ out) {
    __shared__ unsigned short ldsB[64 * LDSK];

    const int tid = threadIdx.x;
    const int wave = tid >> 5;
    const int lane = tid & 31;
    const int lane15 = lane & 15;
    const int lhalf = (lane >> 4) << 3;   // 0 or 8
    const int b = blockIdx.y;
    const int hw0 = blockIdx.x * 64;

    // ---- stage x tile (K=256 x N=64) into LDS as bf16, [N][K] layout ----
    const float* xb = x + (size_t)b * CC * HW + hw0;
#pragma unroll 4
    for (int i = 0; i < 64; i++) {
        int idx = i * 256 + tid;      // 0..16383
        int c = idx >> 6;
        int j = idx & 63;
        ldsB[j * LDSK + c] = f2bf(xb[(size_t)c * HW + j]);
    }
    __syncthreads();

    f32x8 acc[2][4];
#pragma unroll
    for (int mt = 0; mt < 2; mt++)
#pragma unroll
        for (int nt = 0; nt < 4; nt++) acc[mt][nt] = (f32x8)0.0f;

    // ---- K loop: 8 steps of 32, 8 WMMAs per step per wave ----
    for (int k0 = 0; k0 < CC; k0 += 32) {
        Frag afr[2], bfr[4];
#pragma unroll
        for (int mt = 0; mt < 2; mt++) {
            int M = wave * 32 + mt * 16 + lane15;
            const u32x4* p = (const u32x4*)(reswbf + (size_t)M * CC + k0 + lhalf);
            afr[mt].q[0] = p[0];   // K = k0+lhalf .. +7
            afr[mt].q[1] = p[2];   // K = k0+lhalf+16 .. +23
        }
#pragma unroll
        for (int nt = 0; nt < 4; nt++) {
            const unsigned short* row = &ldsB[(nt * 16 + lane15) * LDSK + k0 + lhalf];
            bfr[nt].q[0] = *(const u32x4*)(row);
            bfr[nt].q[1] = *(const u32x4*)(row + 16);
        }
#pragma unroll
        for (int mt = 0; mt < 2; mt++)
#pragma unroll
            for (int nt = 0; nt < 4; nt++)
                acc[mt][nt] = __builtin_amdgcn_wmma_f32_16x16x32_bf16(
                    false, afr[mt].v, false, bfr[nt].v,
                    (short)0, acc[mt][nt], false, false);
    }

    // ---- fused epilogue: out = resconv + res_b + x*softmax(sk) + S/Z ----
    float saj[4];
#pragma unroll
    for (int nt = 0; nt < 4; nt++)
        saj[nt] = __ldg(&sa[(size_t)b * HW + hw0 + nt * 16 + lane15]);

#pragma unroll
    for (int mt = 0; mt < 2; mt++) {
#pragma unroll
        for (int r = 0; r < 8; r++) {
            int c = wave * 32 + mt * 16 + r + lhalf;
            int bc = b * CC + c;
            float cav = __ldg(&caL[bc]);
            float av  = __ldg(&aArr[bc]);
            float iz  = __ldg(&invZ[bc]);
            float soz = __ldg(&SoZ[bc]);
            float rb  = __ldg(&resb[c]);
            size_t rowoff = ((size_t)b * CC + c) * HW + hw0;
#pragma unroll
            for (int nt = 0; nt < 4; nt++) {
                int j = nt * 16 + lane15;
                float s = saj[nt];
                float sk = av * hsig(cav * s) + (1.0f - av) * hsig(cav + s);
                float E = __expf(sk);
                float xv = __ldg(&x[rowoff + j]);
                out[rowoff + j] = acc[mt][nt][r] + rb + xv * E * iz + soz;
            }
        }
    }
}

extern "C" void kernel_launch(void* const* d_in, const int* in_sizes, int n_in,
                              void* d_out, int out_size, void* d_ws, size_t ws_size,
                              hipStream_t stream) {
    const float* x       = (const float*)d_in[0];
    const float* conv0_w = (const float*)d_in[1];
    const float* conv0_b = (const float*)d_in[2];
    const float* bn1_w   = (const float*)d_in[3];
    const float* bn1_b   = (const float*)d_in[4];
    const float* conv1_w = (const float*)d_in[5];
    const float* conv1_b = (const float*)d_in[6];
    const float* ln_g    = (const float*)d_in[7];
    const float* ln_b    = (const float*)d_in[8];
    const float* sk_w1   = (const float*)d_in[9];
    const float* sk_w2   = (const float*)d_in[10];
    const float* res_w   = (const float*)d_in[11];
    const float* res_b   = (const float*)d_in[12];
    float* out = (float*)d_out;

    char* w = (char*)d_ws;
    unsigned short* reswbf = (unsigned short*)w;                 // 131072 B
    float* xmean = (float*)(w + 131072);                         // 8192 f
    float* sa    = xmean + BB * CC;                              // 131072 f
    float* caL   = sa + BB * HW;
    float* u     = caL + BB * CC;
    float* aArr  = u + BB * CC;
    float* invZ  = aArr + BB * CC;
    float* SoZ   = invZ + BB * CC;

    k_cvt_resw<<<CC * CC / 256, 256, 0, stream>>>(res_w, reswbf);
    k_xmean<<<BB * CC, 256, 0, stream>>>(x, xmean);
    k_sa<<<dim3(HW / 256, BB), 256, 0, stream>>>(x, conv1_w, conv1_b, sa);
    k_ca<<<BB, 256, 0, stream>>>(xmean, conv0_w, conv0_b, bn1_w, bn1_b, ln_g, ln_b, caL);
    k_u<<<dim3(CC, BB), 256, 0, stream>>>(sa, caL, u);
    k_sk<<<BB, 256, 0, stream>>>(u, sk_w1, sk_w2, aArr);
    k_zs<<<dim3(CC, BB), 256, 0, stream>>>(x, sa, caL, aArr, invZ, SoZ);
    k_gemm<<<dim3(HW / 64, BB), 256, 0, stream>>>(x, reswbf, res_b, sa, caL, aArr,
                                                  invZ, SoZ, out);
}
